// OuterProductMean_72954314490349
// MI455X (gfx1250) — matrix operand order repaired and checked by reading, not compiled, not run on hardware
//
#include <hip/hip_runtime.h>
#include <hip/hip_bf16.h>

// ---------------------------------------------------------------------------
// OuterProductMean fused kernel for gfx1250 (MI455X), wave32 + bf16 WMMA.
//   stage 1:  z = a^T b   (M=12288, N=12288, K=256) tiled 128x128 per block,
//             B-panel staged in LDS via global_load_async_to_lds_b128
//   stage 2:  out = (z * nmInv) @ Wo + bo, fused through LDS (no z in HBM)
// ---------------------------------------------------------------------------

#define S_LEN 256
#define N_LEN 384
#define C_IN  64
#define C_HID 32
#define C_OUT 128

// LDS B-panel row pitch: 264 halfs = 528 B = 132 dwords; 132 % 64 = 4 banks
// skew per row -> 16 rows x b128 reads cover all 64 banks conflict-free.
#define B_PITCH 264

typedef __attribute__((ext_vector_type(16))) __bf16 v16bf;
typedef __attribute__((ext_vector_type(8)))  __bf16 v8bf;
typedef __attribute__((ext_vector_type(8)))  float  v8f;

union AFrag { v16bf v; v8bf h[2]; };

#define HAS_ASYNC_LDS __has_builtin(__builtin_amdgcn_global_load_async_to_lds_b128)
#if HAS_ASYNC_LDS
typedef int v4i_vs __attribute__((vector_size(16)));          // <4 x i32>
typedef __attribute__((address_space(1))) void   as1_void;
typedef __attribute__((address_space(3))) void   as3_void;
typedef __attribute__((address_space(1))) v4i_vs as1_v4i;
typedef __attribute__((address_space(3))) v4i_vs as3_v4i;
#endif

// ---------------------------------------------------------------------------
// Kernel 1: layernorm + masked projections, stored transposed (K-major, K = s)
//   aT[(n*32+c)*256 + s], bT likewise, bf16.
// One block (64 threads) per (s, n) position.
// ---------------------------------------------------------------------------
__global__ __launch_bounds__(64) void opm_ln_proj(
    const float* __restrict__ m, const int* __restrict__ mask,
    const float* __restrict__ ln_scale, const float* __restrict__ ln_bias,
    const float* __restrict__ Wa, const float* __restrict__ Wb,
    __bf16* __restrict__ aT, __bf16* __restrict__ bT)
{
    const int pos = blockIdx.x;              // s * N_LEN + n
    const int s = pos / N_LEN;
    const int n = pos - s * N_LEN;
    const int t = threadIdx.x;               // 0..63

    __shared__ float red[64];
    __shared__ float mn_sh[64];

    const float x = m[(size_t)pos * C_IN + t];

    // mean
    red[t] = x; __syncthreads();
    #pragma unroll
    for (int off = 32; off > 0; off >>= 1) {
        if (t < off) red[t] += red[t + off];
        __syncthreads();
    }
    const float mu = red[0] * (1.0f / 64.0f);
    __syncthreads();

    // variance (ddof=0)
    const float xm = x - mu;
    red[t] = xm * xm; __syncthreads();
    #pragma unroll
    for (int off = 32; off > 0; off >>= 1) {
        if (t < off) red[t] += red[t + off];
        __syncthreads();
    }
    const float var = red[0] * (1.0f / 64.0f);

    const float mn = xm * rsqrtf(var + 1e-5f) * ln_scale[t] + ln_bias[t];
    mn_sh[t] = mn;
    __syncthreads();

    const float mf = (float)mask[pos];
    const float* W = (t < 32) ? Wa : Wb;
    const int c = t & 31;
    float acc = 0.0f;
    #pragma unroll
    for (int i = 0; i < C_IN; ++i) acc += mn_sh[i] * W[i * C_HID + c];
    acc *= mf;

    const size_t row = (size_t)n * C_HID + c;
    if (t < 32) aT[row * S_LEN + s] = (__bf16)acc;
    else        bT[row * S_LEN + s] = (__bf16)acc;
}

// ---------------------------------------------------------------------------
// Kernel 2: nmInv[i*384+j] = 1 / max(sum_s mask[s,i]*mask[s,j], 1)
// ---------------------------------------------------------------------------
__global__ void opm_nm(const int* __restrict__ mask, float* __restrict__ nmInv)
{
    const int idx = blockIdx.x * blockDim.x + threadIdx.x;
    if (idx >= N_LEN * N_LEN) return;
    const int i = idx / N_LEN;
    const int j = idx - i * N_LEN;
    int acc = 0;
    for (int s = 0; s < S_LEN; ++s)
        acc += mask[s * N_LEN + i] * mask[s * N_LEN + j];
    nmInv[idx] = 1.0f / (float)((acc < 1) ? 1 : acc);
}

// ---------------------------------------------------------------------------
// Kernel 3: WoT[o*1024 + k] = bf16(Wo[k*128 + o])   (B-matrix wants K contiguous)
// ---------------------------------------------------------------------------
__global__ void opm_wot(const float* __restrict__ Wo, __bf16* __restrict__ WoT)
{
    const int idx = blockIdx.x * blockDim.x + threadIdx.x;
    if (idx >= C_OUT * (C_HID * C_HID)) return;
    const int o = idx >> 10;          // /1024
    const int k = idx & 1023;
    WoT[idx] = (__bf16)Wo[(size_t)k * C_OUT + o];
}

// ---------------------------------------------------------------------------
// Kernel 4: fused main. Block = 8 wave32s, owns 4x4 (i,j) pairs = 128x128 z.
// ---------------------------------------------------------------------------
__global__ __launch_bounds__(256) void opm_main(
    const __bf16* __restrict__ aT, const __bf16* __restrict__ bT,
    const __bf16* __restrict__ WoT, const float* __restrict__ nmInv,
    const float* __restrict__ bo, float* __restrict__ out)
{
    const int ib   = blockIdx.x;             // 0..95 (i block of 4)
    const int jb   = blockIdx.y;             // 0..95 (j block of 4)
    const int wave = threadIdx.x >> 5;       // 0..7
    const int lane = threadIdx.x & 31;
    const int lmod = lane & 15;
    const int lhalf = lane >> 4;

    __shared__ __bf16 blds[128 * B_PITCH];   // staged B panel, 67,584 B
    __shared__ __bf16 zlds[16 * 1024];       // z as [pair p][k = c*32+d], 32 KB

    // ---- Stage 0: async-copy B panel (128 rows x 256 halfs) into LDS ------
    // 4096 granules of 16 B; 256 threads x 16 granules each.
    {
        const __bf16* bSrcBase = bT + (size_t)jb * 128 * S_LEN;
        char* bldsB = (char*)blds;
        for (int g = threadIdx.x; g < 4096; g += 256) {
            const int row = g >> 5;
            const int ch  = g & 31;                       // 16-B chunk in row
            const __bf16* src = bSrcBase + row * S_LEN + ch * 8;
            char* dst = bldsB + row * (B_PITCH * 2) + ch * 16;
#if HAS_ASYNC_LDS
            __builtin_amdgcn_global_load_async_to_lds_b128(
                (as1_v4i*)(as1_void*)(void*)const_cast<__bf16*>(src),
                (as3_v4i*)(as3_void*)(void*)dst, 0, 0);
#else
            *(v8bf*)dst = *(const v8bf*)src;
#endif
        }
#if HAS_ASYNC_LDS
#if __has_builtin(__builtin_amdgcn_s_wait_asynccnt)
        __builtin_amdgcn_s_wait_asynccnt(0);
#else
        asm volatile("s_wait_asynccnt 0x0" ::: "memory");
#endif
#endif
        __syncthreads();
    }

    // ---------------- Stage 1: z(128x128) = A(128x256) * B(256x128) --------
    v8f zero = {};
    v8f acc[8];
    #pragma unroll
    for (int t = 0; t < 8; ++t) acc[t] = zero;

    // Wave owns M-tile `wave` (rows wave*16 .. +15 of the (i,c) axis).
    // A 16-bit 16x32 layout: lanes0-15 M=lane, K = kb + {0..7, 16..23};
    //                        lanes16-31 same M, K shifted by +8.
    const __bf16* aRow = aT + ((size_t)ib * 128 + wave * 16 + lmod) * S_LEN;
    const char*   bRowL = (const char*)blds + lmod * (B_PITCH * 2);

    for (int kb = 0; kb < S_LEN; kb += 32) {
        __builtin_prefetch(aRow + kb + 64, 0, 1);   // global_prefetch_b8
        AFrag af;
        af.h[0] = *(const v8bf*)(aRow + kb + 8 * lhalf);
        af.h[1] = *(const v8bf*)(aRow + kb + 8 * lhalf + 16);
        #pragma unroll
        for (int nt = 0; nt < 8; ++nt) {
            // B 16-bit 32x16 layout: lanes0-15 N=lane hold K=kb+{0..7,16..23},
            // lanes16-31 K shifted by +8 -> two 16-B LDS reads.
            const char* bRow = bRowL + nt * 16 * (B_PITCH * 2);
            AFrag bf;
            bf.h[0] = *(const v8bf*)(bRow + (kb + 8 * lhalf) * 2);
            bf.h[1] = *(const v8bf*)(bRow + (kb + 8 * lhalf + 16) * 2);
            acc[nt] = __builtin_amdgcn_wmma_f32_16x16x32_bf16(
                false, af.v, false, bf.v, (short)0, acc[nt], false, false);
        }
    }

    // Scale by 1/num_mask and scatter to LDS as z[p][c*32+d] (bf16).
    // D layout: lane holds (M = r + 8*lhalf, N = lmod) in VGPR r.
    #pragma unroll
    for (int nt = 0; nt < 8; ++nt) {
        const int col = nt * 16 + lmod;            // 0..127 on (j,d) axis
        const int jg  = jb * 4 + (col >> 5);
        #pragma unroll
        for (int r = 0; r < 8; ++r) {
            const int mrow = wave * 16 + r + 8 * lhalf;   // 0..127 on (i,c)
            const int ig   = ib * 4 + (mrow >> 5);
            const float zv = acc[nt][r] * nmInv[ig * N_LEN + jg];
            const int p = ((mrow >> 5) << 2) | (col >> 5);       // pair id
            const int k = ((mrow & 31) << 5) | (col & 31);       // c*32+d
            zlds[p * 1024 + k] = (__bf16)zv;
        }
    }
    __syncthreads();

    // ---------------- Stage 2: out(16x128) = z(16x1024) * Wo(1024x128) -----
    // Wave owns output N-tile `wave` (o = wave*16 .. +15); K loop = 1024.
    v8f acc2 = zero;
    const __bf16* wRow = WoT + ((size_t)wave * 16 + lmod) * 1024;
    const __bf16* zRow = &zlds[lmod * 1024];       // A row m = pair lmod
    for (int ks = 0; ks < 1024; ks += 32) {
        AFrag zf;                                   // ds_load_b128 x2
        zf.h[0] = *(const v8bf*)(zRow + ks + 16 * lhalf);
        zf.h[1] = *(const v8bf*)(zRow + ks + 16 * lhalf + 8);
        const v16bf wf = *(const v16bf*)(wRow + ks + 16 * lhalf);
        acc2 = __builtin_amdgcn_wmma_f32_16x16x32_bf16(
            false, zf.v, false, wf, (short)0, acc2, false, false);
    }

    const int o = wave * 16 + lmod;
    const float bov = bo[o];
    #pragma unroll
    for (int r = 0; r < 8; ++r) {
        const int p  = r + 8 * lhalf;              // pair id 0..15
        const int ig = ib * 4 + (p >> 2);
        const int jg = jb * 4 + (p & 3);
        out[((size_t)ig * N_LEN + jg) * C_OUT + o] = acc2[r] + bov;
    }
}

// ---------------------------------------------------------------------------
// Launcher. Workspace layout (13.44 MB total):
//   aT   bf16 [12288][256]   @ 0          (6,291,456 B)
//   bT   bf16 [12288][256]   @ 6,291,456  (6,291,456 B)
//   WoT  bf16 [128][1024]    @ 12,582,912 (262,144 B)
//   nmInv f32 [384*384]      @ 12,845,056 (589,824 B)
// ---------------------------------------------------------------------------
extern "C" void kernel_launch(void* const* d_in, const int* in_sizes, int n_in,
                              void* d_out, int out_size, void* d_ws, size_t ws_size,
                              hipStream_t stream) {
    const float* m        = (const float*)d_in[0];
    const int*   mask     = (const int*)  d_in[1];
    const float* ln_scale = (const float*)d_in[2];
    const float* ln_bias  = (const float*)d_in[3];
    const float* Wa       = (const float*)d_in[4];
    const float* Wb       = (const float*)d_in[5];
    const float* Wo       = (const float*)d_in[6];
    const float* bo       = (const float*)d_in[7];
    float* out = (float*)d_out;

    char* ws = (char*)d_ws;
    __bf16* aT    = (__bf16*)(ws);
    __bf16* bT    = (__bf16*)(ws + 6291456);
    __bf16* WoT   = (__bf16*)(ws + 12582912);
    float*  nmInv = (float*) (ws + 12845056);

    opm_ln_proj<<<S_LEN * N_LEN, 64, 0, stream>>>(m, mask, ln_scale, ln_bias,
                                                  Wa, Wb, aT, bT);
    opm_nm<<<(N_LEN * N_LEN + 255) / 256, 256, 0, stream>>>(mask, nmInv);
    opm_wot<<<(C_OUT * C_HID * C_HID + 255) / 256, 256, 0, stream>>>(Wo, WoT);
    opm_main<<<dim3(N_LEN / 4, N_LEN / 4), 256, 0, stream>>>(aT, bT, WoT,
                                                             nmInv, bo, out);
}
